// MultiHeadAttention_89172110999812
// MI455X (gfx1250) — matrix operand order repaired and checked by reading, not compiled, not run on hardware
//
#include <hip/hip_runtime.h>
#include <hip/hip_bf16.h>

// ---------------------------------------------------------------------------
// MultiHeadAttention forward for MI455X (gfx1250, wave32, WMMA).
// f16 inputs / f32 accumulate via v_wmma_f32_16x16x32_f16.
// ---------------------------------------------------------------------------

typedef __attribute__((ext_vector_type(16))) _Float16 v16h;
typedef __attribute__((ext_vector_type(8)))  float    v8f;

union F16x16 { v16h v; unsigned u[8]; };

static __device__ inline unsigned pk2(float a, float b) {
    union { _Float16 h[2]; unsigned u; } x;
    x.h[0] = (_Float16)a; x.h[1] = (_Float16)b;
    return x.u;
}

static __device__ inline v8f wmma32(v16h a, v16h b, v8f c) {
    // D = A(16x32 f16) * B(32x16 f16) + C(16x16 f32)
    return __builtin_amdgcn_wmma_f32_16x16x32_f16(
        /*neg_a=*/false, a, /*neg_b=*/false, b,
        /*c_mod=*/(short)0, c, /*reuse_a=*/false, /*reuse_b=*/false);
}

// A fragment (16x32, f16) from LDS tile stored [m][k] (k contiguous, stride in halves).
// Layout (ISA 7.12.2): lanes 0-15 -> M=lane, v0..3: K=0..7, v4..7: K=16..23
//                      lanes16-31 -> M=lane-16, v0..3: K=8..15, v4..7: K=24..31
static __device__ inline v16h load_a_frag(const _Float16* lds, int m_base, int stride) {
    const int L  = threadIdx.x & 31;
    const int m  = m_base + (L & 15);
    const int kb = (L < 16) ? 0 : 4;   // in uint (pair) units: halves 0 or 8
    const unsigned* row = (const unsigned*)(lds + m * stride);
    F16x16 f;
#pragma unroll
    for (int j = 0; j < 4; ++j) f.u[j]     = row[kb + j];
#pragma unroll
    for (int j = 0; j < 4; ++j) f.u[4 + j] = row[8 + kb + j];
    return f.v;
}

// B fragment (32x16, f16) from LDS tile stored [n][k] (k contiguous, stride in halves).
// Layout: lanes 0-15 -> N=lane, v0..3: K=0..7, v4..7: K=16..23; lanes 16-31 -> N=lane-16, v0..3: K=8..15, v4..7: K=24..31.
static __device__ inline v16h load_b_frag(const _Float16* lds, int n_base, int stride) {
    const int L  = threadIdx.x & 31;
    const int n  = n_base + (L & 15);
    const int kh = (L < 16) ? 0 : 4;   // uint units: halves 0 or 8
    const unsigned* row = (const unsigned*)(lds + n * stride);
    F16x16 f;
#pragma unroll
    for (int j = 0; j < 8; ++j) f.u[j] = row[(j < 4) ? (kh + j) : (4 + kh + j)];
    return f.v;
}

// ---------------------------------------------------------------------------
// Kernel 1/3: generic GEMM  C[M,N] = A[M,K] * B[K,N] + bias, f16 WMMA compute.
// 128 threads = 4 waves; block tile 128x128; each wave 64x64 (4x4 WMMA tiles).
// qkv_mode: also scatter cols>=1024 into `present` (B,2,H,S,64) layout.
// ---------------------------------------------------------------------------
__global__ __launch_bounds__(128) void gemm_f16_wmma(
    const float* __restrict__ A, const float* __restrict__ Bm,
    const float* __restrict__ bias, float* __restrict__ Cout,
    float* __restrict__ present, int M, int N, int K, int qkv_mode)
{
    __shared__ __align__(16) _Float16 As[128 * 40];  // [m][k], k padded 32->40
    __shared__ __align__(16) _Float16 Bs[128 * 40];  // [n][k] (transposed), padded

    const int t  = threadIdx.x;
    const int m0 = blockIdx.y * 128;
    const int n0 = blockIdx.x * 128;
    const int w  = t >> 5, L = t & 31;
    const int wr = w >> 1, wc = w & 1;

    v8f acc[4][4] = {};

    for (int k0 = 0; k0 < K; k0 += 32) {
        __syncthreads();
        {   // A tile: thread t owns tile row t, 32 floats -> 16 packed uints
            const float* src = A + (size_t)(m0 + t) * K + k0;
            if (k0 + 32 < K) __builtin_prefetch(src + 32, 0, 3);
            unsigned* dst = (unsigned*)(As + t * 40);
#pragma unroll
            for (int i = 0; i < 16; ++i) dst[i] = pk2(src[2 * i], src[2 * i + 1]);
        }
        {   // B tile: thread t: k-row = t&31, 32 n's starting at (t>>5)*32, transpose into LDS
            const int kr = t & 31, nc = (t >> 5) * 32;
            const float* src = Bm + (size_t)(k0 + kr) * N + n0 + nc;
#pragma unroll
            for (int i = 0; i < 32; ++i) Bs[(nc + i) * 40 + kr] = (_Float16)src[i];
        }
        __syncthreads();

        v16h a[4], b[4];
#pragma unroll
        for (int i = 0; i < 4; ++i) a[i] = load_a_frag(As, wr * 64 + i * 16, 40);
#pragma unroll
        for (int j = 0; j < 4; ++j) b[j] = load_b_frag(Bs, wc * 64 + j * 16, 40);
#pragma unroll
        for (int i = 0; i < 4; ++i)
#pragma unroll
            for (int j = 0; j < 4; ++j)
                acc[i][j] = wmma32(a[i], b[j], acc[i][j]);
    }

    // Epilogue: C layout -> lane = N col, vgpr r = M row (lanes>=16: row+8).
    const int rbase = (L < 16) ? 0 : 8;
#pragma unroll
    for (int i = 0; i < 4; ++i) {
#pragma unroll
        for (int j = 0; j < 4; ++j) {
            const int col = n0 + wc * 64 + j * 16 + (L & 15);
            const float bv = bias[col];
#pragma unroll
            for (int r = 0; r < 8; ++r) {
                const int row = m0 + wr * 64 + i * 16 + rbase + r;
                const float v = acc[i][j][r] + bv;
                Cout[(size_t)row * N + col] = v;
                if (qkv_mode && col >= 1024) {
                    const int kv = (col >= 2048) ? 1 : 0;
                    const int c  = col - (kv ? 2048 : 1024);
                    const int hh = c >> 6, d = c & 63;
                    const int bb = row >> 11, s = row & 2047;
                    present[((((size_t)bb * 2 + kv) * 16 + hh) * 2048 + s) * 64 + d] = v;
                }
            }
        }
    }
}

// ---------------------------------------------------------------------------
// Kernel 2: causal flash attention. Grid (S/64, B*H), 128 threads (4 waves).
// Wave w owns q rows [blk*64 + w*16, +16). K/V streamed in 32-key LDS tiles.
// qkv layout: [b*2048+s][3072] with q at col h*64, k at 1024+h*64, v at 2048+h*64.
// ---------------------------------------------------------------------------
__global__ __launch_bounds__(128) void attn_flash_wmma(
    const float* __restrict__ qkv, float* __restrict__ attn_out)
{
    __shared__ __align__(16) _Float16 Kt[32 * 72];    // [key][d], d padded 64->72
    __shared__ __align__(16) _Float16 VtT[64 * 40];   // [d][key], key padded 32->40
    __shared__ __align__(16) _Float16 Ps[4][16 * 40]; // per-wave P tile [m][k]

    const int t = threadIdx.x, w = t >> 5, L = t & 31;
    const int bh = blockIdx.y;
    const int bb = bh >> 4, h = bh & 15;
    const int qblk = blockIdx.x * 64;
    const int q0 = qblk + w * 16;

    // Q fragments (two K=32 steps over d=64), built straight from global f32.
    v16h qf[2];
    {
        const int m  = L & 15;
        const int kb = (L < 16) ? 0 : 8;
        const size_t rowoff = (size_t)(bb * 2048 + q0 + m) * 3072 + h * 64;
#pragma unroll
        for (int dstep = 0; dstep < 2; ++dstep) {
            const float* src = qkv + rowoff + dstep * 32;
            F16x16 f;
#pragma unroll
            for (int j = 0; j < 8; ++j) {
                const int k = (j < 4) ? (kb + 2 * j) : (16 + kb + 2 * (j - 4));
                f.u[j] = pk2(src[k], src[k + 1]);
            }
            qf[dstep] = f.v;
        }
    }

    v8f oacc[4] = {};
    float mrow[8], lrow[8];
#pragma unroll
    for (int r = 0; r < 8; ++r) { mrow[r] = -1e30f; lrow[r] = 0.f; }

    const int nkt = blockIdx.x * 2 + 2;   // key tiles covering keys <= qblk+63 (uniform per block)
    for (int kt = 0; kt < nkt; ++kt) {
        const int key0 = kt * 32;
        __syncthreads();
        {   // cooperative K/V tile load, f32 -> f16, V transposed
            const int kr = t & 31, dc = (t >> 5) * 16;
            const size_t base = (size_t)(bb * 2048 + key0 + kr) * 3072 + h * 64 + dc;
            const float* ksrc = qkv + base + 1024;
            unsigned* kdst = (unsigned*)(Kt + kr * 72 + dc);
#pragma unroll
            for (int i = 0; i < 8; ++i) kdst[i] = pk2(ksrc[2 * i], ksrc[2 * i + 1]);
            const float* vsrc = qkv + base + 2048;
#pragma unroll
            for (int i = 0; i < 16; ++i) VtT[(dc + i) * 40 + kr] = (_Float16)vsrc[i];
        }
        __syncthreads();

        // scores: two 16-key subtiles, each accumulated over d in two K=32 WMMAs
        v8f sc[2];
#pragma unroll
        for (int st = 0; st < 2; ++st) {
            v8f cacc = {};
#pragma unroll
            for (int dstep = 0; dstep < 2; ++dstep)
                cacc = wmma32(qf[dstep], load_b_frag(Kt + dstep * 32, st * 16, 72), cacc);
            sc[st] = cacc;
        }

        const int n = L & 15;
        const int rbase = (L < 16) ? 0 : 8;
#pragma unroll
        for (int st = 0; st < 2; ++st) {
            const int key = key0 + st * 16 + n;
#pragma unroll
            for (int r = 0; r < 8; ++r) {
                const float s = sc[st][r] * 0.125f;          // 1/sqrt(64)
                sc[st][r] = (key > (q0 + rbase + r)) ? -1e9f : s;
            }
        }

        // online softmax (row lives in the VGPR index -> per-row ops are scalar)
        _Float16* Psw = Ps[w];
#pragma unroll
        for (int r = 0; r < 8; ++r) {
            float mx = fmaxf(sc[0][r], sc[1][r]);
#pragma unroll
            for (int off = 1; off < 16; off <<= 1) mx = fmaxf(mx, __shfl_xor(mx, off, 16));
            const float mn = fmaxf(mrow[r], mx);
            const float alpha = __expf(mrow[r] - mn);
            const float p0 = __expf(sc[0][r] - mn);
            const float p1 = __expf(sc[1][r] - mn);
            float sum = p0 + p1;
#pragma unroll
            for (int off = 1; off < 16; off <<= 1) sum += __shfl_xor(sum, off, 16);
            lrow[r] = lrow[r] * alpha + sum;
            mrow[r] = mn;
            const int prow = rbase + r;
            Psw[prow * 40 + n]      = (_Float16)p0;
            Psw[prow * 40 + 16 + n] = (_Float16)p1;
#pragma unroll
            for (int c = 0; c < 4; ++c) oacc[c][r] = oacc[c][r] * alpha;
        }
        __syncthreads();   // P tile visible; uniform trip count keeps this legal

        // O += P(16x32) * V(32x64)
        const v16h pa = load_a_frag(Psw, 0, 40);
#pragma unroll
        for (int c = 0; c < 4; ++c)
            oacc[c] = wmma32(pa, load_b_frag(VtT, c * 16, 40), oacc[c]);
    }

    // normalize and write merged-head rows [b*2048+q][1024]
    const int n = L & 15;
    const int rbase = (L < 16) ? 0 : 8;
#pragma unroll
    for (int c = 0; c < 4; ++c)
#pragma unroll
        for (int r = 0; r < 8; ++r) {
            const int qrow = q0 + rbase + r;
            attn_out[(size_t)(bb * 2048 + qrow) * 1024 + h * 64 + c * 16 + n] =
                oacc[c][r] / lrow[r];
        }
}

// ---------------------------------------------------------------------------
extern "C" void kernel_launch(void* const* d_in, const int* in_sizes, int n_in,
                              void* d_out, int out_size, void* d_ws, size_t ws_size,
                              hipStream_t stream) {
    const float* x      = (const float*)d_in[0];
    // d_in[1] = mask: causality is applied analytically in-kernel
    const float* w_attn = (const float*)d_in[2];
    const float* b_attn = (const float*)d_in[3];
    const float* w_proj = (const float*)d_in[4];
    const float* b_proj = (const float*)d_in[5];

    float* out     = (float*)d_out;                       // (2,2048,1024)
    float* present = out + (size_t)2 * 2048 * 1024;       // (2,2,16,2048,64)

    float* ws_qkv = (float*)d_ws;                          // 4096 x 3072
    float* ws_att = ws_qkv + (size_t)4096 * 3072;          // 4096 x 1024

    const dim3 blk(128);

    // 1) QKV = x @ w_attn + b_attn  (also scatters K/V into `present`)
    gemm_f16_wmma<<<dim3(3072 / 128, 4096 / 128), blk, 0, stream>>>(
        x, w_attn, b_attn, ws_qkv, present, 4096, 3072, 1024, 1);

    // 2) causal flash attention per (b,h), 64-row q blocks
    attn_flash_wmma<<<dim3(2048 / 64, 32), blk, 0, stream>>>(ws_qkv, ws_att);

    // 3) out = attn_out @ w_proj + b_proj
    gemm_f16_wmma<<<dim3(1024 / 128, 4096 / 128), blk, 0, stream>>>(
        ws_att, w_proj, b_proj, out, nullptr, 4096, 1024, 1024, 0);
}
